// TransformerDocumentQuantizer_89867895701687
// MI455X (gfx1250) — matrix operand order repaired and checked by reading, not compiled, not run on hardware
//
#include <hip/hip_runtime.h>
#include <stdint.h>

// ---------------------------------------------------------------------------
// VQ-VAE vector quantization for MI455X (gfx1250), wave32 + WMMA + TDM.
// inputs  d_in[0]: [64,2048,256] f32   (N=131072 rows, D=256)
// codebook d_in[1]: [1024,256]   f32   (K=1024)
// d_out: quantized [N*256] f32 | indices [N] f32 | loss | perplexity
// ---------------------------------------------------------------------------

typedef __attribute__((ext_vector_type(16))) _Float16 v16h;
typedef __attribute__((ext_vector_type(8)))  _Float16 v8h;
typedef __attribute__((ext_vector_type(4)))  _Float16 v4h;
typedef __attribute__((ext_vector_type(8)))  float    v8f;
typedef __attribute__((ext_vector_type(4)))  unsigned u32x4;
typedef __attribute__((ext_vector_type(8)))  int      i32x8;
typedef __attribute__((ext_vector_type(4)))  int      i32x4;

#define D_DIM     256
#define K_CODES   1024
#define N_VEC     (64 * 2048)
#define KCHUNK    32                       // codes staged per LDS buffer
#define N_STAGES  (K_CODES / KCHUNK)       // 32
#define LDS_ROW_H 264                      // halves per LDS row: 256 + 8 pad (16B) -> conflict-free b128
#define ROW_TILES 2                        // 16-row A tiles per wave (32 rows/wave)
#define ROWS_PER_BLOCK (ROW_TILES * 16 * 8) // 256 rows per 8-wave block

#if defined(__HIP_DEVICE_COMPILE__) && \
    __has_builtin(__builtin_amdgcn_tensor_load_to_lds) && \
    __has_builtin(__builtin_amdgcn_s_wait_tensorcnt)
#define VQ_USE_TDM 1
#else
#define VQ_USE_TDM 0
#endif

#if VQ_USE_TDM
// Issue a TDM load of KCHUNK x 256 f16 codebook rows into (padded) LDS.
__device__ __forceinline__ void tdm_issue(const _Float16* gsrc, unsigned lds_addr) {
    unsigned long long ga = (unsigned long long)(uintptr_t)gsrc;
    // Group0: count=1 | lds_addr | global_addr[56:0] | type=2
    u32x4 g0 = { 1u,
                 lds_addr,
                 (unsigned)(ga & 0xffffffffull),
                 (unsigned)(((ga >> 32) & 0x01ffffffull) | 0x80000000u) };
    // Group1: data_size=2B, pad_enable, pad_interval=6 (128 dw = 512B), pad_amount=3 (4 dw = 16B)
    //         tensor_dim0=256, tensor_dim1=KCHUNK, tile_dim0=256, tile_dim1=KCHUNK, dim0_stride=256
    i32x8 g1 = { (int)0x07910000u,                // pad/datasize control, mask=0 (not in cluster)
                 (int)(256u << 16),               // tensor_dim0 lo in [31:16]
                 (int)(((unsigned)KCHUNK) << 16), // tensor_dim1 lo in [31:16]
                 (int)(256u << 16),               // tile_dim0 in [31:16]
                 (int)KCHUNK,                     // tile_dim1 in [15:0], tile_dim2=0
                 256,                             // tensor_dim0_stride[31:0]
                 0, 0 };
    i32x4 g2 = {0, 0, 0, 0};
    i32x4 g3 = {0, 0, 0, 0};
#if defined(__clang_major__) && (__clang_major__ >= 23)
    i32x8 gz = {0, 0, 0, 0, 0, 0, 0, 0};
    __builtin_amdgcn_tensor_load_to_lds(g0, g1, g2, g3, gz, 0);
#else
    __builtin_amdgcn_tensor_load_to_lds(g0, g1, g2, g3, 0);
#endif
}
#endif

// ---------------------------------------------------------------------------
// Prep: codebook f32 -> f16 copy + exact fp32 row norms.  grid: K_CODES x 64
// ---------------------------------------------------------------------------
__global__ __launch_bounds__(64)
void vq_prep(const float* __restrict__ cb, _Float16* __restrict__ cb16,
             float* __restrict__ cnorm) {
    const int row = blockIdx.x;
    const int t   = threadIdx.x;
    float4 v = ((const float4*)(cb + (size_t)row * D_DIM))[t];
    v4h h;
    h[0] = (_Float16)v.x; h[1] = (_Float16)v.y;
    h[2] = (_Float16)v.z; h[3] = (_Float16)v.w;
    ((v4h*)(cb16 + (size_t)row * D_DIM))[t] = h;
    float s = v.x * v.x + v.y * v.y + v.z * v.z + v.w * v.w;
    #pragma unroll
    for (int off = 16; off >= 1; off >>= 1) s += __shfl_xor(s, off, 32);
    __shared__ float tmp[2];
    if ((t & 31) == 0) tmp[t >> 5] = s;
    __syncthreads();
    if (t == 0) cnorm[row] = tmp[0] + tmp[1];
}

// ---------------------------------------------------------------------------
// Zero histogram + loss accumulator (must be re-zeroed every call)
// ---------------------------------------------------------------------------
__global__ void vq_zero(float* __restrict__ hist, float* __restrict__ lossAcc) {
    int i = blockIdx.x * blockDim.x + threadIdx.x;
    if (i < K_CODES) hist[i] = 0.0f;
    if (i == 0) lossAcc[0] = 0.0f;
}

// ---------------------------------------------------------------------------
// Main fused kernel: WMMA GEMM + argmin + gather + loss + histogram.
// grid: N_VEC/256 blocks x 256 threads; each wave owns 2 x 16-row tiles.
// ---------------------------------------------------------------------------
__global__ __launch_bounds__(256)
void vq_main(const float* __restrict__ x, const float* __restrict__ cb,
             const _Float16* __restrict__ cb16, const float* __restrict__ cnorm,
             float* __restrict__ qout, float* __restrict__ idxout,
             float* __restrict__ hist, float* __restrict__ lossAcc) {
    __shared__ __align__(16) _Float16 smem[2][KCHUNK * LDS_ROW_H];

    const int tid     = threadIdx.x;
    const int lane    = tid & 31;
    const int wave    = tid >> 5;
    const int hh      = lane >> 4;       // lane half (0/1)
    const int l16     = lane & 15;
    const int rowBase = blockIdx.x * ROWS_PER_BLOCK + wave * (ROW_TILES * 16);

    // ---- A fragments: ROW_TILES x (16 rows x D=256) of x, f32 -> f16 ----
    // 16-bit A 16x32 layout: lane holds row M=l16; elems 0..7 = K hh*8+0..7,
    // elems 8..15 = K 16+hh*8+0..7 (per 32-wide K slice s).
    v16h a[ROW_TILES][8];
    #pragma unroll
    for (int tt = 0; tt < ROW_TILES; ++tt) {
        const float* xr = x + (size_t)(rowBase + tt * 16 + l16) * D_DIM;
        #pragma unroll
        for (int s = 0; s < 8; ++s) {
            const float* p = xr + s * 32 + hh * 8;
            float4 f0 = *(const float4*)(p);
            float4 f1 = *(const float4*)(p + 4);
            float4 f2 = *(const float4*)(p + 16);
            float4 f3 = *(const float4*)(p + 20);
            v16h t;
            t[0]  = (_Float16)f0.x; t[1]  = (_Float16)f0.y; t[2]  = (_Float16)f0.z; t[3]  = (_Float16)f0.w;
            t[4]  = (_Float16)f1.x; t[5]  = (_Float16)f1.y; t[6]  = (_Float16)f1.z; t[7]  = (_Float16)f1.w;
            t[8]  = (_Float16)f2.x; t[9]  = (_Float16)f2.y; t[10] = (_Float16)f2.z; t[11] = (_Float16)f2.w;
            t[12] = (_Float16)f3.x; t[13] = (_Float16)f3.y; t[14] = (_Float16)f3.z; t[15] = (_Float16)f3.w;
            a[tt][s] = t;
        }
    }

    // ---- running argmin state (per lane: 8 rows of its half, its column slot) ----
    float minv[ROW_TILES][8];
    int   mini[ROW_TILES][8];
    #pragma unroll
    for (int tt = 0; tt < ROW_TILES; ++tt)
        #pragma unroll
        for (int r = 0; r < 8; ++r) { minv[tt][r] = 3.402823e38f; mini[tt][r] = 0; }

#if VQ_USE_TDM
    if (wave == 0)
        tdm_issue(cb16, (unsigned)(uintptr_t)(void*)&smem[0][0]);
#endif

    for (int t = 0; t < N_STAGES; ++t) {
        const int cur = t & 1;
#if VQ_USE_TDM
        if (wave == 0) {
            if (t + 1 < N_STAGES) {
                tdm_issue(cb16 + (size_t)(t + 1) * KCHUNK * D_DIM,
                          (unsigned)(uintptr_t)(void*)&smem[cur ^ 1][0]);
                __builtin_amdgcn_s_wait_tensorcnt(1);   // stage t done; t+1 in flight
            } else {
                __builtin_amdgcn_s_wait_tensorcnt(0);
            }
        }
        __syncthreads();
#else
        __syncthreads();
        {   // cooperative staging with the same padded layout
            const _Float16* gsrc = cb16 + (size_t)t * KCHUNK * D_DIM;
            for (int c = tid; c < KCHUNK * D_DIM / 8; c += 256) {
                int row = c >> 5;
                int col = (c & 31) << 3;
                *(v8h*)&smem[cur][row * LDS_ROW_H + col] =
                    *(const v8h*)(gsrc + (size_t)row * D_DIM + col);
            }
        }
        __syncthreads();
#endif
        // ---- compute score tiles: dot(x_row, code) via WMMA.
        //      Each B fragment feeds ROW_TILES WMMAs (2 independent chains). ----
        #pragma unroll
        for (int j = 0; j < KCHUNK / 16; ++j) {
            v8f acc[ROW_TILES];
            #pragma unroll
            for (int tt = 0; tt < ROW_TILES; ++tt)
                acc[tt] = (v8f){0.f, 0.f, 0.f, 0.f, 0.f, 0.f, 0.f, 0.f};
            // B 32x16 layout: lane holds col N=l16; elems 0..7 = K hh*8+0..7, elems 8..15 = K 16+hh*8+0..7
            const _Float16* brow = &smem[cur][(j * 16 + l16) * LDS_ROW_H + hh * 8];
            #pragma unroll
            for (int s = 0; s < 8; ++s) {
                v8h lo = *(const v8h*)(brow + s * 32);
                v8h hi = *(const v8h*)(brow + s * 32 + 16);
                v16h bfrag = __builtin_shufflevector(lo, hi,
                    0, 1, 2, 3, 4, 5, 6, 7, 8, 9, 10, 11, 12, 13, 14, 15);
                #pragma unroll
                for (int tt = 0; tt < ROW_TILES; ++tt)
                    acc[tt] = __builtin_amdgcn_wmma_f32_16x16x32_f16(
                        false, a[tt][s], false, bfrag, (short)0, acc[tt], false, false);
            }
            const int kc = t * KCHUNK + j * 16 + l16;   // this lane's code column
            const float cn = cnorm[kc];
            #pragma unroll
            for (int tt = 0; tt < ROW_TILES; ++tt)
                #pragma unroll
                for (int r = 0; r < 8; ++r) {
                    float dist = cn - 2.0f * acc[tt][r];   // + ||x||^2 is row-constant
                    if (dist < minv[tt][r]) { minv[tt][r] = dist; mini[tt][r] = kc; }
                }
        }
#if VQ_USE_TDM
        __syncthreads();   // everyone done with smem[cur] before TDM overwrites it
#endif
    }

    // ---- per tile: reduce argmin, write indices/hist, gather + loss ----
    float lacc = 0.0f;
    #pragma unroll
    for (int tt = 0; tt < ROW_TILES; ++tt) {
        const int tileBase = rowBase + tt * 16;
        // reduce argmin across the 16 lanes sharing each row half
        #pragma unroll
        for (int r = 0; r < 8; ++r) {
            #pragma unroll
            for (int off = 8; off >= 1; off >>= 1) {
                float ov = __shfl_xor(minv[tt][r], off, 32);
                int   oi = __shfl_xor(mini[tt][r], off, 32);
                if (ov < minv[tt][r] || (ov == minv[tt][r] && oi < mini[tt][r])) {
                    minv[tt][r] = ov; mini[tt][r] = oi;
                }
            }
        }
        // write indices (as float) + histogram (lanes 0 and 16)
        if (l16 == 0) {
            #pragma unroll
            for (int r = 0; r < 8; ++r) {
                int gr = tileBase + hh * 8 + r;
                idxout[gr] = (float)mini[tt][r];
                atomicAdd(&hist[mini[tt][r]], 1.0f);
            }
        }
        // gather fp32 codebook rows -> quantized out, accumulate loss
        #pragma unroll
        for (int m = 0; m < 16; ++m) {
            const int src = (m < 8) ? 0 : 16;
            const int idx = __shfl(mini[tt][m & 7], src, 32);
            const int gr  = tileBase + m;
            const float4* crow = (const float4*)(cb + (size_t)idx * D_DIM);
            const float4* xrow = (const float4*)(x + (size_t)gr * D_DIM);
            float4* qrow = (float4*)(qout + (size_t)gr * D_DIM);
            #pragma unroll
            for (int h2 = 0; h2 < 2; ++h2) {
                int o = h2 * 32 + lane;
                float4 c  = crow[o];
                float4 xv = xrow[o];
                qrow[o] = c;
                float dx = c.x - xv.x, dy = c.y - xv.y;
                float dz = c.z - xv.z, dw = c.w - xv.w;
                lacc += dx * dx + dy * dy + dz * dz + dw * dw;
            }
        }
    }
    #pragma unroll
    for (int off = 16; off >= 1; off >>= 1) lacc += __shfl_xor(lacc, off, 32);
    if (lane == 0) atomicAdd(lossAcc, lacc);
}

// ---------------------------------------------------------------------------
// Final scalars: loss = 1.25 * MSE ; perplexity = exp(-sum p log(p+1e-10))
// ---------------------------------------------------------------------------
__global__ __launch_bounds__(256)
void vq_final(const float* __restrict__ hist, const float* __restrict__ lossAcc,
              float* __restrict__ out) {
    __shared__ float sh[8];
    const int t = threadIdx.x;
    float acc = 0.0f;
    for (int k = t; k < K_CODES; k += 256) {
        float p = hist[k] * (1.0f / (float)N_VEC);
        acc += p * logf(p + 1e-10f);
    }
    #pragma unroll
    for (int off = 16; off >= 1; off >>= 1) acc += __shfl_xor(acc, off, 32);
    if ((t & 31) == 0) sh[t >> 5] = acc;
    __syncthreads();
    if (t == 0) {
        float tot = 0.0f;
        #pragma unroll
        for (int w = 0; w < 8; ++w) tot += sh[w];
        const size_t QN = (size_t)N_VEC * D_DIM;
        out[QN + N_VEC]     = 1.25f * lossAcc[0] / (float)((size_t)N_VEC * D_DIM);
        out[QN + N_VEC + 1] = expf(-tot);
    }
}

// ---------------------------------------------------------------------------
extern "C" void kernel_launch(void* const* d_in, const int* in_sizes, int n_in,
                              void* d_out, int out_size, void* d_ws, size_t ws_size,
                              hipStream_t stream) {
    (void)in_sizes; (void)n_in; (void)out_size; (void)ws_size;
    const float* x  = (const float*)d_in[0];
    const float* cb = (const float*)d_in[1];

    // workspace layout (needs ~533 KB)
    char* ws = (char*)d_ws;
    _Float16* cb16  = (_Float16*)ws;                 // 1024*256*2 = 524288 B
    float* cnorm    = (float*)(ws + 524288);         // 4096 B
    float* hist     = (float*)(ws + 528384);         // 4096 B
    float* lossAcc  = (float*)(ws + 532480);         // 16 B

    float* out    = (float*)d_out;
    float* qout   = out;
    float* idxout = out + (size_t)N_VEC * D_DIM;

    vq_zero<<<4, 256, 0, stream>>>(hist, lossAcc);
    vq_prep<<<K_CODES, 64, 0, stream>>>(cb, cb16, cnorm);
    vq_main<<<N_VEC / ROWS_PER_BLOCK, 256, 0, stream>>>(x, cb, cb16, cnorm,
                                                        qout, idxout, hist, lossAcc);
    vq_final<<<1, 256, 0, stream>>>(hist, lossAcc, out);
}
